// GraphPDHGNet_69157563400860
// MI455X (gfx1250) — matrix operand order repaired and checked by reading, not compiled, not run on hardware
//
#include <hip/hip_runtime.h>
#include <hip/hip_bf16.h>
#include <math.h>

typedef __attribute__((ext_vector_type(16))) __bf16 v16bf;
typedef __attribute__((ext_vector_type(8)))  __bf16 bf16x8;
typedef __attribute__((ext_vector_type(8)))  float  v8f;

#define LAM_ 1.0f

__device__ __forceinline__ v8f wmma_bf16(v16bf a, v16bf b, v8f c) {
  // 8 args: (neg_a, A, neg_b, B, c_mod, C, reuse_a, reuse_b)
  return __builtin_amdgcn_wmma_f32_16x16x32_bf16(false, a, false, b, (short)0, c,
                                                 false, false);
}

__device__ __forceinline__ void fill8_bf(v16bf& a, int base, const __bf16* p) {
  bf16x8 v = *(const bf16x8*)p;
#pragma unroll
  for (int i = 0; i < 8; ++i) a[base + i] = v[i];
}

__device__ __forceinline__ bf16x8 cvt8(float4 u0, float4 u1) {
  bf16x8 v;
  v[0] = (__bf16)u0.x; v[1] = (__bf16)u0.y; v[2] = (__bf16)u0.z; v[3] = (__bf16)u0.w;
  v[4] = (__bf16)u1.x; v[5] = (__bf16)u1.y; v[6] = (__bf16)u1.z; v[7] = (__bf16)u1.w;
  return v;
}

// ---------------------------------------------------------------------------
// Weight prep: concat-K transposed bf16 weights + folded biases.
// ---------------------------------------------------------------------------
__global__ void prep_kernel(const float* __restrict__ W_up,  const float* __restrict__ b_up,
                            const float* __restrict__ W_agg, const float* __restrict__ b_agg,
                            const float* __restrict__ W_nf,  const float* __restrict__ b_nf,
                            const float* __restrict__ W_res, const float* __restrict__ b_res,
                            const float* __restrict__ W_aggl,const float* __restrict__ b_aggl,
                            const float* __restrict__ W_n1,  const float* __restrict__ W_n2,
                            int nd, int ed, int od,
                            __bf16* __restrict__ Wedge_t, float* __restrict__ bedge,
                            __bf16* __restrict__ Wnode_t, float* __restrict__ bnode,
                            __bf16* __restrict__ Wn1_t,   __bf16* __restrict__ Wn2_t)
{
  int idx = blockIdx.x * blockDim.x + threadIdx.x;
  const int Ke = ed + nd;
  const int Kn = nd + 64 + od;
  const int n1 = od * Ke, n2 = od * Kn, n3 = od * od;
  if (idx < n1) {
    int n = idx / Ke, k = idx % Ke;
    float v = (k < ed) ? W_up[(size_t)k * od + n] : W_agg[(size_t)(k - ed) * od + n];
    Wedge_t[idx] = (__bf16)v;
    return;
  }
  idx -= n1;
  if (idx < n2) {
    int n = idx / Kn, k = idx % Kn;
    float v;
    if (k < nd)           v = W_nf [(size_t)k * od + n];
    else if (k < nd + 64) v = W_res[(size_t)(k - nd) * od + n];
    else                  v = W_aggl[(size_t)(k - nd - 64) * od + n];
    Wnode_t[idx] = (__bf16)v;
    return;
  }
  idx -= n2;
  if (idx < n3) {
    int n = idx / od, k = idx % od;
    Wn1_t[idx] = (__bf16)W_n1[(size_t)k * od + n];
    return;
  }
  idx -= n3;
  if (idx < n3) {
    int n = idx / od, k = idx % od;
    Wn2_t[idx] = (__bf16)W_n2[(size_t)k * od + n];
    return;
  }
  idx -= n3;
  if (idx < od) {
    bedge[idx] = b_up[idx] + b_agg[idx];
    bnode[idx] = b_nf[idx] + b_res[idx] + b_aggl[idx];
  }
}

__global__ void cvt_f32_bf16(const float* __restrict__ in, __bf16* __restrict__ out,
                             long long n)
{
  long long i = (long long)blockIdx.x * blockDim.x + threadIdx.x;
  if (i < n) out[i] = (__bf16)in[i];
}

__global__ void zero_kernel(float* __restrict__ p, long long n)
{
  long long i = (long long)blockIdx.x * blockDim.x + threadIdx.x;
  if (i < n) p[i] = 0.0f;
}

// ---------------------------------------------------------------------------
// Edge kernel: block = 64 edges x full OD (OD/16 waves; 4 row-tiles per wave so
// each B fragment feeds 4 WMMAs; all 4 A fragments preloaded to overlap LDS
// latency with matrix issue).
// Phase 1 (cooperative): stage A = [e_row | sqrtw*(h[src]-h[dst])] as bf16 in LDS.
// Phase 2: fully unrolled bf16 WMMA over K; bias; row-norm via ds_add_f32;
//          l2-ball projection (v_rsq); store e_proj; scatter sqrtw*e_proj to agg.
// ---------------------------------------------------------------------------
template<int ED, int ND, int OD, bool FINAL>
__global__ __launch_bounds__(32 * (OD / 16))
void edge_kernel(const __bf16* __restrict__ e_in,
                 const float* __restrict__ h_in,
                 const long long* __restrict__ src,
                 const long long* __restrict__ dst,
                 const float* __restrict__ w,
                 const __bf16* __restrict__ Wt,
                 const float* __restrict__ bias,
                 __bf16* __restrict__ e_out_bf,
                 float* __restrict__ e_out_f32,
                 float* __restrict__ agg)
{
  constexpr int K    = ED + ND;
  constexpr int NT   = 32 * (OD / 16);  // threads per block
  constexpr int MT   = 4;               // row tiles per wave
  constexpr int ROWS = 16 * MT;         // 64 edges per block
  constexpr int TPR  = NT / ROWS;       // threads cooperating on one row

  __shared__ __bf16    sA[ROWS * K];
  __shared__ float     s_norm2[ROWS];
  __shared__ float     s_sw[ROWS];
  __shared__ long long s_src[ROWS];
  __shared__ long long s_dst[ROWS];

  const int lane = threadIdx.x & 31;
  const int wv   = threadIdx.x >> 5;
  const int hi   = lane >> 4;
  const int l15  = lane & 15;
  const long long ebase = (long long)blockIdx.x * ROWS;

  if (threadIdx.x < ROWS) {
    s_norm2[threadIdx.x] = 0.0f;
    s_sw[threadIdx.x]    = sqrtf(w[ebase + threadIdx.x]);
    s_src[threadIdx.x]   = src[ebase + threadIdx.x];
    s_dst[threadIdx.x]   = dst[ebase + threadIdx.x];
  }
  __syncthreads();

  // ---- phase 1: cooperative A staging (ROWS x K bf16) ----
  {
    const int r  = threadIdx.x / TPR;
    const int li = threadIdx.x % TPR;
    const long long rowg = ebase + r;
    const float sw = s_sw[r];
    const size_t hs = (size_t)s_src[r] * ND;
    const size_t hd = (size_t)s_dst[r] * ND;
#pragma unroll
    for (int j = li; j < K / 8; j += TPR) {
      const int k8 = j * 8;
      bf16x8 v;
      if (k8 < ED) {
        v = *(const bf16x8*)(e_in + (size_t)rowg * ED + k8);
      } else {
        const float* ps = h_in + hs + (k8 - ED);
        const float* pd = h_in + hd + (k8 - ED);
        float4 s0 = *(const float4*)ps, s1 = *(const float4*)(ps + 4);
        float4 d0 = *(const float4*)pd, d1 = *(const float4*)(pd + 4);
        float4 u0, u1;
        u0.x = sw * (s0.x - d0.x); u0.y = sw * (s0.y - d0.y);
        u0.z = sw * (s0.z - d0.z); u0.w = sw * (s0.w - d0.w);
        u1.x = sw * (s1.x - d1.x); u1.y = sw * (s1.y - d1.y);
        u1.z = sw * (s1.z - d1.z); u1.w = sw * (s1.w - d1.w);
        v = cvt8(u0, u1);
      }
      *(bf16x8*)&sA[r * K + k8] = v;
    }
  }
  __syncthreads();

  // ---- phase 2: WMMA over K; preload all A fragments, then 4 back-to-back WMMAs ----
  v8f acc[MT];
#pragma unroll
  for (int mt = 0; mt < MT; ++mt) acc[mt] = (v8f){};

#pragma unroll
  for (int ks = 0; ks < K / 32; ++ks) {
    v16bf b;
    const __bf16* bp = Wt + (size_t)(wv * 16 + l15) * K + (ks * 32 + hi * 8);
    fill8_bf(b, 0, bp);
    fill8_bf(b, 8, bp + 16);
    v16bf a[MT];
#pragma unroll
    for (int mt = 0; mt < MT; ++mt) {
      const __bf16* ap = &sA[(mt * 16 + l15) * K + ks * 32 + hi * 8];
      fill8_bf(a[mt], 0, ap);
      fill8_bf(a[mt], 8, ap + 16);
    }
#pragma unroll
    for (int mt = 0; mt < MT; ++mt)
      acc[mt] = wmma_bf16(a[mt], b, acc[mt]);
  }

  const int ncol = wv * 16 + l15;
  const float bn = bias[ncol];
#pragma unroll
  for (int mt = 0; mt < MT; ++mt) {
#pragma unroll
    for (int v = 0; v < 8; ++v) {
      acc[mt][v] += bn;
      atomicAdd(&s_norm2[mt * 16 + v + hi * 8], acc[mt][v] * acc[mt][v]);
    }
  }
  __syncthreads();

#pragma unroll
  for (int mt = 0; mt < MT; ++mt) {
#pragma unroll
    for (int v = 0; v < 8; ++v) {
      const int m = mt * 16 + v + hi * 8;     // D layout: row = VGPR + 8*hi
      // scale = min(1, r / max(norm, eps))  ==  min(1, r * rsqrt(max(norm2, eps^2)))
      const float rr    = LAM_ * s_sw[m];
      const float scale = fminf(1.0f, rr * __frsqrt_rn(fmaxf(s_norm2[m], 1e-24f)));
      const float val   = acc[mt][v] * scale;
      const long long gr = ebase + m;
      if constexpr (FINAL) e_out_f32[(size_t)gr * OD + ncol] = val;
      else                 e_out_bf [(size_t)gr * OD + ncol] = (__bf16)val;
      unsafeAtomicAdd(&agg[(size_t)s_dst[m] * OD + ncol], s_sw[m] * val);
    }
  }
}

// ---------------------------------------------------------------------------
// Node kernel: block = 16 nodes x full OD. Three chained bf16 WMMA GEMMs.
//   sA <- [h | x | agg] (bf16, cooperative)         (K1 = ND + 64 + OD)
//   ni  = sA @ Wnode^T + bnode        -> sB
//   hid = silu(sB @ Wn1 + b1)         -> sA (reuse)
//   h'  = sA @ Wn2 + b2               -> global
// ---------------------------------------------------------------------------
template<int ND, int OD>
__global__ __launch_bounds__(32 * (OD / 16))
void node_kernel(const float* __restrict__ h_in,
                 const float* __restrict__ x,
                 const float* __restrict__ agg,
                 const __bf16* __restrict__ Wnt, const float* __restrict__ bnode,
                 const __bf16* __restrict__ Wn1t, const float* __restrict__ bn1,
                 const __bf16* __restrict__ Wn2t, const float* __restrict__ bn2,
                 float* __restrict__ h_out)
{
  constexpr int K1  = ND + 64 + OD;
  constexpr int NT  = 32 * (OD / 16);
  constexpr int TPR = NT / 16;

  __shared__ __bf16 sA[16 * K1];
  __shared__ __bf16 sB[16 * OD];

  const int lane = threadIdx.x & 31;
  const int wv   = threadIdx.x >> 5;
  const int hi   = lane >> 4;
  const int l15  = lane & 15;
  const long long nbase = (long long)blockIdx.x * 16;
  const int ncol = wv * 16 + l15;

  // ---- cooperative staging of A = [h | x | agg] as bf16 ----
  {
    const int r  = threadIdx.x / TPR;
    const int li = threadIdx.x % TPR;
    const long long rowg = nbase + r;
#pragma unroll
    for (int j = li; j < K1 / 8; j += TPR) {
      const int k8 = j * 8;
      const float* p;
      if (k8 < ND)           p = h_in + (size_t)rowg * ND + k8;
      else if (k8 < ND + 64) p = x    + (size_t)rowg * 64 + (k8 - ND);
      else                   p = agg  + (size_t)rowg * OD + (k8 - ND - 64);
      float4 u0 = *(const float4*)p;
      float4 u1 = *(const float4*)(p + 4);
      *(bf16x8*)&sA[r * K1 + k8] = cvt8(u0, u1);
    }
  }
  __syncthreads();

  // ---- stage 1: ni = A @ Wnode^T + bnode -> sB ----
  v8f acc = {};
#pragma unroll
  for (int ks = 0; ks < K1 / 32; ++ks) {
    v16bf a, b;
    const __bf16* ap = &sA[l15 * K1 + ks * 32 + hi * 8];
    fill8_bf(a, 0, ap);
    fill8_bf(a, 8, ap + 16);
    const __bf16* bp = Wnt + (size_t)ncol * K1 + (ks * 32 + hi * 8);
    fill8_bf(b, 0, bp);
    fill8_bf(b, 8, bp + 16);
    acc = wmma_bf16(a, b, acc);
  }
  {
    const float bn = bnode[ncol];
#pragma unroll
    for (int v = 0; v < 8; ++v)
      sB[(v + hi * 8) * OD + ncol] = (__bf16)(acc[v] + bn);
  }
  __syncthreads();

  // ---- stage 2: hid = silu(ni @ Wn1 + b1) -> sA (reused) ----
  v8f acc2 = {};
#pragma unroll
  for (int ks = 0; ks < OD / 32; ++ks) {
    v16bf a, b;
    const __bf16* ap = &sB[l15 * OD + ks * 32 + hi * 8];
    fill8_bf(a, 0, ap);
    fill8_bf(a, 8, ap + 16);
    const __bf16* bp = Wn1t + (size_t)ncol * OD + (ks * 32 + hi * 8);
    fill8_bf(b, 0, bp);
    fill8_bf(b, 8, bp + 16);
    acc2 = wmma_bf16(a, b, acc2);
  }
  {
    const float bn = bn1[ncol];
#pragma unroll
    for (int v = 0; v < 8; ++v) {
      float t = acc2[v] + bn;
      acc2[v] = t / (1.0f + __expf(-t));   // silu (v_exp_f32 path)
    }
  }
  __syncthreads();
#pragma unroll
  for (int v = 0; v < 8; ++v)
    sA[(v + hi * 8) * OD + ncol] = (__bf16)acc2[v];
  __syncthreads();

  // ---- stage 3: h' = hid @ Wn2 + b2 ----
  v8f acc3 = {};
#pragma unroll
  for (int ks = 0; ks < OD / 32; ++ks) {
    v16bf a, b;
    const __bf16* ap = &sA[l15 * OD + ks * 32 + hi * 8];
    fill8_bf(a, 0, ap);
    fill8_bf(a, 8, ap + 16);
    const __bf16* bp = Wn2t + (size_t)ncol * OD + (ks * 32 + hi * 8);
    fill8_bf(b, 0, bp);
    fill8_bf(b, 8, bp + 16);
    acc3 = wmma_bf16(a, b, acc3);
  }
  {
    const float bn = bn2[ncol];
#pragma unroll
    for (int v = 0; v < 8; ++v)
      h_out[(size_t)(nbase + v + hi * 8) * OD + ncol] = acc3[v] + bn;
  }
}

// ---------------------------------------------------------------------------
extern "C" void kernel_launch(void* const* d_in, const int* in_sizes, int n_in,
                              void* d_out, int out_size, void* d_ws, size_t ws_size,
                              hipStream_t stream)
{
  (void)n_in; (void)out_size; (void)ws_size;
  const int DIN = 64, DHID = 128;

  const float*     h0   = (const float*)d_in[0];
  const float*     e0   = (const float*)d_in[1];
  const long long* eidx = (const long long*)d_in[2];   // int64 edge_index [2,E]
  const float*     w    = (const float*)d_in[3];
  const float*     x    = (const float*)d_in[4];

  const int N = in_sizes[0] / DIN;
  const int E = in_sizes[3];
  const long long* src = eidx;
  const long long* dst = eidx + E;

  // workspace carve-out (256B aligned slabs)
  char* base = (char*)d_ws;
  size_t off = 0;
  auto carve = [&](size_t bytes) -> void* {
    off = (off + 255) & ~(size_t)255;
    void* p = base + off;
    off += bytes;
    return p;
  };
  __bf16* eb0   = (__bf16*)carve((size_t)E * DHID * sizeof(__bf16));
  __bf16* eb1   = (__bf16*)carve((size_t)E * DHID * sizeof(__bf16));
  float*  hb0   = (float*) carve((size_t)N * DHID * sizeof(float));
  float*  hb1   = (float*) carve((size_t)N * DHID * sizeof(float));
  float*  agg   = (float*) carve((size_t)N * DHID * sizeof(float));
  __bf16* Wedge = (__bf16*)carve((size_t)DHID * 256 * sizeof(__bf16));
  float*  bedge = (float*) carve((size_t)DHID * sizeof(float));
  __bf16* Wnode = (__bf16*)carve((size_t)DHID * 320 * sizeof(__bf16));
  float*  bnode = (float*) carve((size_t)DHID * sizeof(float));
  __bf16* Wn1t  = (__bf16*)carve((size_t)DHID * DHID * sizeof(__bf16));
  __bf16* Wn2t  = (__bf16*)carve((size_t)DHID * DHID * sizeof(__bf16));

  static const int nds[4] = {64, 128, 128, 128};
  static const int eds[4] = {64, 128, 128, 128};
  static const int ods[4] = {128, 128, 128, 64};

  // input e (fp32) -> bf16 working buffer
  {
    long long n = (long long)E * DIN;
    cvt_f32_bf16<<<(unsigned)((n + 255) / 256), 256, 0, stream>>>(e0, eb0, n);
  }

  const float*  h_cur = h0;
  const __bf16* e_cur = eb0;
  __bf16* ebufs[2] = {eb1, eb0};
  float*  hbufs[2] = {hb0, hb1};

  for (int L = 0; L < 4; ++L) {
    const int nd = nds[L], ed = eds[L], od = ods[L];
    const float* W_up   = (const float*)d_in[5 + 14 * L + 0];
    const float* b_up   = (const float*)d_in[5 + 14 * L + 1];
    const float* W_agg  = (const float*)d_in[5 + 14 * L + 2];
    const float* b_agg  = (const float*)d_in[5 + 14 * L + 3];
    const float* W_nf   = (const float*)d_in[5 + 14 * L + 4];
    const float* b_nf   = (const float*)d_in[5 + 14 * L + 5];
    const float* W_res  = (const float*)d_in[5 + 14 * L + 6];
    const float* b_res  = (const float*)d_in[5 + 14 * L + 7];
    const float* W_aggl = (const float*)d_in[5 + 14 * L + 8];
    const float* b_aggl = (const float*)d_in[5 + 14 * L + 9];
    const float* W_n1   = (const float*)d_in[5 + 14 * L + 10];
    const float* b_n1   = (const float*)d_in[5 + 14 * L + 11];
    const float* W_n2   = (const float*)d_in[5 + 14 * L + 12];
    const float* b_n2   = (const float*)d_in[5 + 14 * L + 13];

    const int Ke = ed + nd, Kn = nd + 64 + od;
    const int total = od * Ke + od * Kn + 2 * od * od + od;
    prep_kernel<<<(total + 255) / 256, 256, 0, stream>>>(
        W_up, b_up, W_agg, b_agg, W_nf, b_nf, W_res, b_res, W_aggl, b_aggl,
        W_n1, W_n2, nd, ed, od, Wedge, bedge, Wnode, bnode, Wn1t, Wn2t);

    {
      long long n = (long long)N * od;
      zero_kernel<<<(unsigned)((n + 255) / 256), 256, 0, stream>>>(agg, n);
    }

    __bf16* e_next   = ebufs[L & 1];
    float* e_out_f32 = (float*)d_out + (size_t)N * DIN;   // e slot of output tuple
    float* h_next    = (L == 3) ? (float*)d_out : hbufs[L & 1];

    const unsigned egrid = (unsigned)(E / 64);   // 64 edges per block
    const unsigned ngrid = (unsigned)(N / 16);

    switch (L) {
      case 0:
        edge_kernel<64, 64, 128, false><<<egrid, 256, 0, stream>>>(
            e_cur, h_cur, src, dst, w, Wedge, bedge, e_next, e_out_f32, agg);
        node_kernel<64, 128><<<ngrid, 256, 0, stream>>>(
            h_cur, x, agg, Wnode, bnode, Wn1t, b_n1, Wn2t, b_n2, h_next);
        break;
      case 1:
        edge_kernel<128, 128, 128, false><<<egrid, 256, 0, stream>>>(
            e_cur, h_cur, src, dst, w, Wedge, bedge, e_next, e_out_f32, agg);
        node_kernel<128, 128><<<ngrid, 256, 0, stream>>>(
            h_cur, x, agg, Wnode, bnode, Wn1t, b_n1, Wn2t, b_n2, h_next);
        break;
      case 2:
        edge_kernel<128, 128, 128, false><<<egrid, 256, 0, stream>>>(
            e_cur, h_cur, src, dst, w, Wedge, bedge, e_next, e_out_f32, agg);
        node_kernel<128, 128><<<ngrid, 256, 0, stream>>>(
            h_cur, x, agg, Wnode, bnode, Wn1t, b_n1, Wn2t, b_n2, h_next);
        break;
      default:
        edge_kernel<128, 128, 64, true><<<egrid, 128, 0, stream>>>(
            e_cur, h_cur, src, dst, w, Wedge, bedge, e_next, e_out_f32, agg);
        node_kernel<128, 64><<<ngrid, 128, 0, stream>>>(
            h_cur, x, agg, Wnode, bnode, Wn1t, b_n1, Wn2t, b_n2, h_next);
        break;
    }

    e_cur = e_next;
    h_cur = h_next;
  }
}
